// LSTMTagger_72172630442176
// MI455X (gfx1250) — matrix operand order repaired and checked by reading, not compiled, not run on hardware
//
#include <hip/hip_runtime.h>
#include <cstdint>

typedef __attribute__((ext_vector_type(16))) _Float16 v16h;
typedef __attribute__((ext_vector_type(8)))  float    v8f;

// ---------------------------------------------------------------------------
// WMMA per-lane layout maps (CDNA5 ISA, 16-bit 16x32 A / 32x16 B)
// A: lanes 0-15 -> M=0..15, elems 0..7 -> K=0..7, elems 8..15 -> K=16..23;
//    lanes 16-31 same M, K offset by 8 (K=8..15 / 24..31).
// B: lanes 0-15 -> N=0..15, same K mapping as A (K=0..7 / 16..23; lanes 16-31 K=8..15 / 24..31).
// C/D f32: lane (n + 16*(mhalf)), vgpr r -> M = mhalf*8 + r.
// ---------------------------------------------------------------------------
__device__ __forceinline__ int kmap_a(int lane, int e) {
    return ((lane >> 4) << 3) + e + (e & 8);
}
__device__ __forceinline__ int kmap_b(int lane, int e) {
    return ((lane >> 4) << 3) + e + (e & 8);
}
__device__ __forceinline__ float sigf(float x) {
    return 1.0f / (1.0f + __expf(-x));
}

// ============================= packing kernels =============================
// Pack weight W[N,K] (row major) into B-tiles (tile idx = nt*Ktiles+kt).
__global__ void pack_b(const float* __restrict__ W, _Float16* __restrict__ dst,
                       int N, int K, int Ktiles, int tiles_total) {
    int g = (blockIdx.x * blockDim.x + threadIdx.x) >> 5;
    if (g >= tiles_total) return;
    int lane = threadIdx.x & 31;
    int nt = g / Ktiles, kt = g % Ktiles;
    _Float16* out = dst + (size_t)g * 512 + lane * 16;
    int n = nt * 16 + (lane & 15);
#pragma unroll
    for (int e = 0; e < 16; ++e) {
        int k = kt * 32 + kmap_b(lane, e);
        float v = (n < N && k < K) ? W[(size_t)n * K + k] : 0.0f;
        out[e] = (_Float16)v;
    }
}

// Pack a plain row-major f32 matrix src[M x K] (row stride ld) into A-tiles.
__global__ void pack_a_plain(const float* __restrict__ src, _Float16* __restrict__ dst,
                             int M, int K, int ld, int Ktiles, int tiles_total) {
    int g = (blockIdx.x * blockDim.x + threadIdx.x) >> 5;
    if (g >= tiles_total) return;
    int lane = threadIdx.x & 31;
    int mt = g / Ktiles, kt = g % Ktiles;
    _Float16* out = dst + (size_t)g * 512 + lane * 16;
    int m = mt * 16 + (lane & 15);
#pragma unroll
    for (int e = 0; e < 16; ++e) {
        int k = kt * 32 + kmap_a(lane, e);
        float v = (m < M && k < K) ? src[(size_t)m * ld + k] : 0.0f;
        out[e] = (_Float16)v;
    }
}

// char embeddings gathered + packed: m = s*16+t, row = char_emb[char_idx[s,t]]
__global__ void pack_a_ce(const int* __restrict__ char_idx,
                          const float* __restrict__ char_emb,
                          _Float16* __restrict__ dst) {
    int g = (blockIdx.x * blockDim.x + threadIdx.x) >> 5; // 300*4 tiles
    if (g >= 1200) return;
    int lane = threadIdx.x & 31;
    int mt = g >> 2, kt = g & 3;
    _Float16* out = dst + (size_t)g * 512 + lane * 16;
    int m = mt * 16 + (lane & 15);            // 0..4799, always valid
    int ci = char_idx[m];
    const float* row = char_emb + (size_t)ci * 100;
#pragma unroll
    for (int e = 0; e < 16; ++e) {
        int k = kt * 32 + kmap_a(lane, e);
        out[e] = (_Float16)((k < 100) ? row[k] : 0.0f);
    }
}

// last valid char per word: row s = char_emb[char_idx[s, lens[s]-1]]
__global__ void pack_a_xlast(const int* __restrict__ char_idx,
                             const int* __restrict__ char_lens,
                             const float* __restrict__ char_emb,
                             _Float16* __restrict__ dst) {
    int g = (blockIdx.x * blockDim.x + threadIdx.x) >> 5; // 19*4 tiles
    if (g >= 76) return;
    int lane = threadIdx.x & 31;
    int mt = g >> 2, kt = g & 3;
    _Float16* out = dst + (size_t)g * 512 + lane * 16;
    int m = mt * 16 + (lane & 15);
    const float* row = nullptr;
    if (m < 300) {
        int ci = char_idx[m * 16 + (char_lens[m] - 1)];
        row = char_emb + (size_t)ci * 100;
    }
#pragma unroll
    for (int e = 0; e < 16; ++e) {
        int k = kt * 32 + kmap_a(lane, e);
        float v = (row && k < 100) ? row[k] : 0.0f;
        out[e] = (_Float16)v;
    }
}

// sentence feature rows: [we+pos (128) | hf (100) | hb (100)] = K 328, Kt=11
__global__ void pack_a_xs(const int* __restrict__ word_idx,
                          const float* __restrict__ word_emb,
                          const float* __restrict__ pos_emb,
                          const float* __restrict__ fchar,
                          _Float16* __restrict__ dst) {
    int g = (blockIdx.x * blockDim.x + threadIdx.x) >> 5; // 19*11 tiles
    if (g >= 209) return;
    int lane = threadIdx.x & 31;
    int mt = g / 11, kt = g % 11;
    _Float16* out = dst + (size_t)g * 512 + lane * 16;
    int s = mt * 16 + (lane & 15);
    int wi = (s < 300) ? word_idx[s] : 0;
#pragma unroll
    for (int e = 0; e < 16; ++e) {
        int k = kt * 32 + kmap_a(lane, e);
        float v = 0.0f;
        if (s < 300) {
            if (k < 128)       v = word_emb[(size_t)wi * 128 + k] + pos_emb[(size_t)s * 128 + k];
            else if (k < 328)  v = fchar[(size_t)s * 200 + (k - 128)]; // hf then hb, contiguous
        }
        out[e] = (_Float16)v;
    }
}

// ============================= generic WMMA GEMM ===========================
// One wave per 16x16 output tile; C[m,n] = sum_k A[m,k]*B[n,k]^T (+ bias[n]).
__global__ void gemm_wmma(const _Float16* __restrict__ A, const _Float16* __restrict__ B,
                          const float* __restrict__ bias, float* __restrict__ C,
                          int Mtiles, int Ntiles, int Ktiles, int M, int N, int ldc) {
    int g = (blockIdx.x * blockDim.x + threadIdx.x) >> 5;
    if (g >= Mtiles * Ntiles) return;
    int lane = threadIdx.x & 31;
    int mt = g / Ntiles, nt = g % Ntiles;
    v8f acc = {};
    for (int kt = 0; kt < Ktiles; ++kt) {
        v16h a = *(const v16h*)(A + ((size_t)(mt * Ktiles + kt)) * 512 + lane * 16);
        v16h b = *(const v16h*)(B + ((size_t)(nt * Ktiles + kt)) * 512 + lane * 16);
        acc = __builtin_amdgcn_wmma_f32_16x16x32_f16(false, a, false, b,
                                                     (short)0, acc, false, false);
    }
    int n = nt * 16 + (lane & 15);
    if (n < N) {
        float bv = bias ? bias[n] : 0.0f;
        union { v8f v; float f[8]; } u; u.v = acc;
#pragma unroll
        for (int r = 0; r < 8; ++r) {
            int m = mt * 16 + ((lane >> 4) << 3) + r;
            if (m < M) C[(size_t)m * ldc + n] = u.f[r] + bv;
        }
    }
}

// ====================== char forward LSTM (16 steps) =======================
// One block = 16 words, 25 waves (one per 16 gate columns). Per step:
// gates = XW(precomputed) + bias + h @ Whh^T (WMMA, A repacked in LDS).
__global__ __launch_bounds__(800) void char_fwd_scan(
        const float* __restrict__ XW, const _Float16* __restrict__ Bhh,
        const float* __restrict__ bias, const float* __restrict__ h0c,
        const float* __restrict__ c0c, const int* __restrict__ char_lens,
        float* __restrict__ fchar) {
    __shared__ _Float16 aT[4 * 512];      // h packed as A-tiles (K padded to 128)
    __shared__ float gb[16 * 400];        // gates for 16 words
    __shared__ float hb[16 * 100];
    __shared__ float cb[16 * 100];
    __shared__ int   ln[16];
    int tid = threadIdx.x;
    int s0 = blockIdx.x * 16;

    for (int i = tid; i < 1600; i += 800) {
        int w = i / 100, u = i % 100; int s = s0 + w;
        hb[i] = (s < 300) ? h0c[(size_t)s * 200 + u] : 0.0f;  // h0c[:,0,:]
        cb[i] = (s < 300) ? c0c[(size_t)s * 200 + u] : 0.0f;
    }
    if (tid < 16) { int s = s0 + tid; ln[tid] = (s < 300) ? char_lens[s] : 0; }
    __syncthreads();
    for (int i = tid; i < 2048; i += 800) {   // initial A-pack of h
        int kt = i >> 9, r = i & 511, lane = r >> 4, e = r & 15;
        int w = lane & 15; int k = kt * 32 + kmap_a(lane, e);
        aT[i] = (k < 100) ? (_Float16)hb[w * 100 + k] : (_Float16)0.0f;
    }
    __syncthreads();

    int wave = tid >> 5, lane = tid & 31;
    int n = wave * 16 + (lane & 15);          // gate column 0..399
    v16h bw0 = *(const v16h*)(Bhh + (size_t)(wave * 4 + 0) * 512 + lane * 16);
    v16h bw1 = *(const v16h*)(Bhh + (size_t)(wave * 4 + 1) * 512 + lane * 16);
    v16h bw2 = *(const v16h*)(Bhh + (size_t)(wave * 4 + 2) * 512 + lane * 16);
    v16h bw3 = *(const v16h*)(Bhh + (size_t)(wave * 4 + 3) * 512 + lane * 16);
    float bn = bias[n];

    for (int t = 0; t < 16; ++t) {
        union { v8f v; float f[8]; } acc;
#pragma unroll
        for (int r = 0; r < 8; ++r) {         // C = XW + bias
            int w = ((lane >> 4) << 3) + r; int s = s0 + w;
            acc.f[r] = ((s < 300) ? XW[(size_t)(s * 16 + t) * 400 + n] : 0.0f) + bn;
        }
        v16h a0 = *(const v16h*)(aT + 0 * 512 + lane * 16);
        v16h a1 = *(const v16h*)(aT + 1 * 512 + lane * 16);
        v16h a2 = *(const v16h*)(aT + 2 * 512 + lane * 16);
        v16h a3 = *(const v16h*)(aT + 3 * 512 + lane * 16);
        acc.v = __builtin_amdgcn_wmma_f32_16x16x32_f16(false, a0, false, bw0, (short)0, acc.v, false, false);
        acc.v = __builtin_amdgcn_wmma_f32_16x16x32_f16(false, a1, false, bw1, (short)0, acc.v, false, false);
        acc.v = __builtin_amdgcn_wmma_f32_16x16x32_f16(false, a2, false, bw2, (short)0, acc.v, false, false);
        acc.v = __builtin_amdgcn_wmma_f32_16x16x32_f16(false, a3, false, bw3, (short)0, acc.v, false, false);
#pragma unroll
        for (int r = 0; r < 8; ++r) {
            int w = ((lane >> 4) << 3) + r;
            gb[w * 400 + n] = acc.f[r];
        }
        __syncthreads();
        for (int i = tid; i < 1600; i += 800) {   // cell update (masked by len)
            int w = i / 100, u = i % 100;
            if (t < ln[w]) {
                float gi = gb[w * 400 + u],        gf = gb[w * 400 + 100 + u];
                float gg = gb[w * 400 + 200 + u],  go = gb[w * 400 + 300 + u];
                float cn = sigf(gf) * cb[i] + sigf(gi) * tanhf(gg);
                float hn = sigf(go) * tanhf(cn);
                cb[i] = cn; hb[i] = hn;
            }
        }
        __syncthreads();
        for (int i = tid; i < 2048; i += 800) {   // repack h for next step
            int kt = i >> 9, r = i & 511, ll = r >> 4, e = r & 15;
            int w = ll & 15; int k = kt * 32 + kmap_a(ll, e);
            aT[i] = (k < 100) ? (_Float16)hb[w * 100 + k] : (_Float16)0.0f;
        }
        __syncthreads();
    }
    for (int i = tid; i < 1600; i += 800) {
        int w = i / 100, u = i % 100; int s = s0 + w;
        if (s < 300) fchar[(size_t)s * 200 + u] = hb[i];
    }
}

// ===================== char backward single cell ===========================
__global__ void char_bwd_cell(const float* __restrict__ Gx, const float* __restrict__ Gh,
                              const float* __restrict__ b, const float* __restrict__ c0c,
                              float* __restrict__ fchar) {
    int i = blockIdx.x * blockDim.x + threadIdx.x;
    if (i >= 30000) return;
    int s = i / 100, u = i % 100;
    const float* gx = Gx + (size_t)s * 400;
    const float* gh = Gh + (size_t)s * 400;
    float gi = gx[u]       + gh[u]       + b[u];
    float gf = gx[100 + u] + gh[100 + u] + b[100 + u];
    float gg = gx[200 + u] + gh[200 + u] + b[200 + u];
    float go = gx[300 + u] + gh[300 + u] + b[300 + u];
    float cn = sigf(gf) * c0c[(size_t)s * 200 + 100 + u] + sigf(gi) * tanhf(gg);
    fchar[(size_t)s * 200 + 100 + u] = sigf(go) * tanhf(cn);
}

// ==================== sentence BiLSTM sequential scan ======================
// 2 blocks (fwd/bwd), 512 threads. Whh row per thread in registers, h in LDS.
__global__ __launch_bounds__(512) void sent_scan(
        const float* __restrict__ Gf, const float* __restrict__ Gb,
        const float* __restrict__ Whh_f, const float* __restrict__ Whh_b,
        const float* __restrict__ h0s, const float* __restrict__ c0s,
        float* __restrict__ outc) {
    __shared__ float hs[100], cs[100], gbf[400];
    int dir = blockIdx.x;
    const float* G   = dir ? Gb : Gf;
    const float* Whh = dir ? Whh_b : Whh_f;
    int tid = threadIdx.x;
    float w[100];
    if (tid < 400) {
#pragma unroll
        for (int k = 0; k < 100; ++k) w[k] = Whh[(size_t)tid * 100 + k];
    }
    if (tid < 100) { hs[tid] = h0s[dir * 100 + tid]; cs[tid] = c0s[dir * 100 + tid]; }
    __syncthreads();
    for (int step = 0; step < 300; ++step) {
        int trow = dir ? (299 - step) : step;
        if (tid < 400) {
            float g = G[(size_t)trow * 400 + tid];   // already includes bias
#pragma unroll
            for (int k = 0; k < 100; ++k) g += w[k] * hs[k];
            gbf[tid] = g;
        }
        if (tid == 0 && step + 1 < 300) {
            int nrow = dir ? (298 - step) : (step + 1);
            __builtin_prefetch(G + (size_t)nrow * 400, 0, 3);
        }
        __syncthreads();
        if (tid < 100) {
            float cn = sigf(gbf[100 + tid]) * cs[tid] + sigf(gbf[tid]) * tanhf(gbf[200 + tid]);
            float hn = sigf(gbf[300 + tid]) * tanhf(cn);
            cs[tid] = cn; hs[tid] = hn;
            outc[(size_t)trow * 200 + dir * 100 + tid] = hn;
        }
        __syncthreads();
    }
}

// ================================ host side ================================
extern "C" void kernel_launch(void* const* d_in, const int* in_sizes, int n_in,
                              void* d_out, int out_size, void* d_ws, size_t ws_size,
                              hipStream_t stream) {
    (void)in_sizes; (void)n_in; (void)out_size; (void)ws_size;
    const int*   word_idx  = (const int*)  d_in[0];
    const int*   char_idx  = (const int*)  d_in[1];
    const int*   char_lens = (const int*)  d_in[2];
    const float* word_emb  = (const float*)d_in[3];
    const float* pos_emb   = (const float*)d_in[4];
    const float* char_emb  = (const float*)d_in[5];
    const float* c_Wih_f   = (const float*)d_in[6];
    const float* c_Whh_f   = (const float*)d_in[7];
    const float* c_b_f     = (const float*)d_in[8];
    const float* c_Wih_b   = (const float*)d_in[9];
    const float* c_Whh_b   = (const float*)d_in[10];
    const float* c_b_b     = (const float*)d_in[11];
    const float* s_Wih_f   = (const float*)d_in[12];
    const float* s_Whh_f   = (const float*)d_in[13];
    const float* s_b_f     = (const float*)d_in[14];
    const float* s_Wih_b   = (const float*)d_in[15];
    const float* s_Whh_b   = (const float*)d_in[16];
    const float* s_b_b     = (const float*)d_in[17];
    const float* W_tag     = (const float*)d_in[18];
    const float* b_tag     = (const float*)d_in[19];
    const float* W_freq    = (const float*)d_in[20];
    const float* b_freq    = (const float*)d_in[21];
    const float* h0c       = (const float*)d_in[22];
    const float* c0c       = (const float*)d_in[23];
    const float* h0s       = (const float*)d_in[24];
    const float* c0s       = (const float*)d_in[25];
    float* out = (float*)d_out;

    char* ws = (char*)d_ws;
    size_t off = 0;
    auto alloc = [&](size_t bytes) -> char* {
        char* p = ws + off; off = (off + bytes + 255) & ~(size_t)255; return p;
    };
    // packed B matrices (tiles of 512 halves)
    _Float16* B_cih_f = (_Float16*)alloc(100 * 1024);
    _Float16* B_cih_b = (_Float16*)alloc(100 * 1024);
    _Float16* B_chh_f = (_Float16*)alloc(100 * 1024);
    _Float16* B_chh_b = (_Float16*)alloc(100 * 1024);
    _Float16* B_sih_f = (_Float16*)alloc(275 * 1024);
    _Float16* B_sih_b = (_Float16*)alloc(275 * 1024);
    _Float16* B_tagp  = (_Float16*)alloc(28 * 1024);
    _Float16* B_freqp = (_Float16*)alloc(7 * 1024);
    // packed A matrices
    _Float16* A_ce  = (_Float16*)alloc(1200 * 1024);
    _Float16* A_xl  = (_Float16*)alloc(76 * 1024);
    _Float16* A_h0  = (_Float16*)alloc(76 * 1024);
    _Float16* A_xs  = (_Float16*)alloc(209 * 1024);
    _Float16* A_out = (_Float16*)alloc(133 * 1024);
    // f32 intermediates
    float* XW_f  = (float*)alloc(4800 * 400 * 4);
    float* Gbx   = (float*)alloc(300 * 400 * 4);
    float* Gbh   = (float*)alloc(300 * 400 * 4);
    float* fchar = (float*)alloc(300 * 200 * 4);
    float* G_f   = (float*)alloc(300 * 400 * 4);
    float* G_b   = (float*)alloc(300 * 400 * 4);
    float* outc  = (float*)alloc(300 * 200 * 4);

    const int TPB = 256;
    auto tb = [](int tiles) { return dim3((tiles + 7) / 8); };
    auto gb = [](int waves) { return dim3((waves + 7) / 8); };

    // 1) pack weights
    pack_b<<<tb(100), TPB, 0, stream>>>(c_Wih_f, B_cih_f, 400, 100, 4, 100);
    pack_b<<<tb(100), TPB, 0, stream>>>(c_Wih_b, B_cih_b, 400, 100, 4, 100);
    pack_b<<<tb(100), TPB, 0, stream>>>(c_Whh_f, B_chh_f, 400, 100, 4, 100);
    pack_b<<<tb(100), TPB, 0, stream>>>(c_Whh_b, B_chh_b, 400, 100, 4, 100);
    pack_b<<<tb(275), TPB, 0, stream>>>(s_Wih_f, B_sih_f, 400, 328, 11, 275);
    pack_b<<<tb(275), TPB, 0, stream>>>(s_Wih_b, B_sih_b, 400, 328, 11, 275);
    pack_b<<<tb(28),  TPB, 0, stream>>>(W_tag,  B_tagp,  50, 200, 7, 28);
    pack_b<<<tb(7),   TPB, 0, stream>>>(W_freq, B_freqp, 10, 200, 7, 7);

    // 2) pack activations known up-front
    pack_a_ce   <<<tb(1200), TPB, 0, stream>>>(char_idx, char_emb, A_ce);
    pack_a_xlast<<<tb(76),   TPB, 0, stream>>>(char_idx, char_lens, char_emb, A_xl);
    pack_a_plain<<<tb(76),   TPB, 0, stream>>>(h0c + 100, A_h0, 300, 100, 200, 4, 76);

    // 3) input-projection GEMMs (WMMA)
    gemm_wmma<<<gb(300 * 25), TPB, 0, stream>>>(A_ce, B_cih_f, nullptr, XW_f,
                                                300, 25, 4, 4800, 400, 400);
    gemm_wmma<<<gb(19 * 25),  TPB, 0, stream>>>(A_xl, B_cih_b, nullptr, Gbx,
                                                19, 25, 4, 300, 400, 400);
    gemm_wmma<<<gb(19 * 25),  TPB, 0, stream>>>(A_h0, B_chh_b, nullptr, Gbh,
                                                19, 25, 4, 300, 400, 400);

    // 4) char LSTM forward scan (WMMA recurrence, 16 words per block)
    char_fwd_scan<<<19, 800, 0, stream>>>(XW_f, B_chh_f, c_b_f, h0c, c0c,
                                          char_lens, fchar);
    // 5) char backward single cell
    char_bwd_cell<<<(30000 + TPB - 1) / TPB, TPB, 0, stream>>>(Gbx, Gbh, c_b_b, c0c, fchar);

    // 6) sentence features -> packed A, pre-gates GEMMs (bias folded in)
    pack_a_xs<<<tb(209), TPB, 0, stream>>>(word_idx, word_emb, pos_emb, fchar, A_xs);
    gemm_wmma<<<gb(19 * 25), TPB, 0, stream>>>(A_xs, B_sih_f, s_b_f, G_f,
                                               19, 25, 11, 300, 400, 400);
    gemm_wmma<<<gb(19 * 25), TPB, 0, stream>>>(A_xs, B_sih_b, s_b_b, G_b,
                                               19, 25, 11, 300, 400, 400);

    // 7) sequential bidirectional scan (latency critical path)
    sent_scan<<<2, 512, 0, stream>>>(G_f, G_b, s_Whh_f, s_Whh_b, h0s, c0s, outc);

    // 8) heads
    pack_a_plain<<<tb(133), TPB, 0, stream>>>(outc, A_out, 300, 200, 200, 7, 133);
    gemm_wmma<<<gb(19 * 4), TPB, 0, stream>>>(A_out, B_tagp, b_tag, out,
                                              19, 4, 7, 300, 50, 50);
    gemm_wmma<<<gb(19 * 1), TPB, 0, stream>>>(A_out, B_freqp, b_freq, out + 15000,
                                              19, 1, 7, 300, 10, 10);
}
